// RNN_cls_st_user_11897059410484
// MI455X (gfx1250) — matrix operand order repaired and checked by reading, not compiled, not run on hardware
//
#include <hip/hip_runtime.h>
#include <math.h>

#define S_DIM 100
#define U_DIM 128
#define H_DIM 256
#define V_DIM 10000
#define H3    (3*H_DIM)   // 768
#define H2    (2*H_DIM)   // 512

typedef __attribute__((ext_vector_type(16))) __bf16 bf16x16;
typedef __attribute__((ext_vector_type(8)))  float  f32x8;
typedef __attribute__((ext_vector_type(4)))  unsigned int u32x4;
typedef __attribute__((ext_vector_type(8)))  unsigned int u32x8;

// ---------------- workspace layout (bytes, all multiples of 256) ----------------
static constexpr size_t OFF_ENC = 0;                                        // V*H bf16
static constexpr size_t OFF_WIH = OFF_ENC + (size_t)V_DIM*H_DIM*2;          // 3H*H bf16
static constexpr size_t OFF_WHH = OFF_WIH + (size_t)H3*H_DIM*2;
static constexpr size_t OFF_FCW = OFF_WHH + (size_t)H3*H_DIM*2;             // V*2H bf16
static constexpr size_t OFF_GI  = OFF_FCW + (size_t)V_DIM*H2*2;             // S*U*3H f32
static constexpr size_t OFF_GH  = OFF_GI  + (size_t)S_DIM*U_DIM*H3*4;      // U*3H f32
static constexpr size_t OFF_OUT = OFF_GH  + (size_t)U_DIM*H3*4;             // S*U*H f32
static constexpr size_t OFF_HPF = OFF_OUT + (size_t)S_DIM*U_DIM*H_DIM*4;   // U*H f32
static constexpr size_t OFF_HPB = OFF_HPF + (size_t)U_DIM*H_DIM*4;          // U*H bf16
static constexpr size_t OFF_OPU = OFF_HPB + (size_t)U_DIM*H_DIM*2;          // S*U*2H bf16

__device__ __forceinline__ unsigned short f2bf(float f) {
  unsigned u = __float_as_uint(f);
  unsigned r = u + 0x7FFFu + ((u >> 16) & 1u);   // round-to-nearest-even
  return (unsigned short)(r >> 16);
}

// A fragment: 16x32 bf16 tile, row-major source. k order used here (not the ISA table order):
// lanes 0-15 hold K={0..15}, lanes 16-31 K={16..31}, contiguous (matches B fragment).
__device__ __forceinline__ bf16x16 load_a_frag(const __bf16* row, int lane, int k0) {
  const int khalf = lane >> 4;
  bf16x16 a;
#pragma unroll
  for (int v = 0; v < 8; ++v) {
    const int k = k0 + (khalf << 4) + (v << 1);
    a[2*v]   = row[k];
    a[2*v+1] = row[k+1];
  }
  return a;
}

// B fragment (32x16 of W^T) from a row-major [rows=N][K contiguous] source.
__device__ __forceinline__ bf16x16 load_b_frag(const __bf16* W, int ldw, int n0, int lane, int k0) {
  const int n    = n0 + (lane & 15);
  const int koff = k0 + ((lane >> 4) << 4);
  const __bf16* p = W + (size_t)n * ldw + koff;
  bf16x16 b;
#pragma unroll
  for (int e = 0; e < 16; ++e) b[e] = p[e];
  return b;
}

// Issue a 2D Tensor-Data-Mover load: tile_d1 rows x tile_d0 elems (2B each)
// from row-major global (row stride stride_d0 elems) into LDS at lds_off.
// D# built per CDNA5 ISA. Tracked by TENSORcnt.
__device__ __forceinline__ void tdm_load_2d(unsigned lds_off, const void* gptr,
                                            unsigned tensor_d0, unsigned tensor_d1,
                                            unsigned tile_d0, unsigned tile_d1,
                                            unsigned stride_d0) {
  const unsigned long long ga = (unsigned long long)gptr;
  u32x4 g0;
  g0[0] = 1u;                                            // count=1, user descriptor
  g0[1] = lds_off;                                       // lds_addr (bytes)
  g0[2] = (unsigned)(ga & 0xffffffffu);                  // global_addr[31:0]
  g0[3] = (unsigned)((ga >> 32) & 0x1ffffffu) | (2u << 30); // addr[56:32] | type=2
  u32x8 g1;
  g1[0] = 1u << 16;                                      // data_size=1 (2 bytes)
  g1[1] = (tensor_d0 & 0xffffu) << 16;                   // tensor_dim0[15:0]
  g1[2] = (tensor_d0 >> 16) | ((tensor_d1 & 0xffffu) << 16); // dim0 hi | dim1 lo
  g1[3] = (tensor_d1 >> 16) | (tile_d0 << 16);           // dim1 hi | tile_dim0
  g1[4] = tile_d1;                                       // tile_dim1 (tile_dim2=0)
  g1[5] = stride_d0;                                     // tensor_dim0_stride lo32
  g1[6] = 0;
  g1[7] = 0;
  asm volatile("tensor_load_to_lds %0, %1" :: "s"(g0), "s"(g1) : "memory");
}

// ---------------- stage 0: convert weights f32 -> bf16 ----------------
__global__ void k_convert(const float* enc_W, const float* W_ih, const float* W_hh,
                          const float* fc_W, unsigned short* enc_b, unsigned short* wih_b,
                          unsigned short* whh_b, unsigned short* fcw_b) {
  const long N_ENC = (long)V_DIM * H_DIM;
  const long N_W   = (long)H3 * H_DIM;
  const long N_FC  = (long)V_DIM * H2;
  const long total = N_ENC + 2*N_W + N_FC;
  for (long i = (long)blockIdx.x * blockDim.x + threadIdx.x; i < total;
       i += (long)gridDim.x * blockDim.x) {
    if (i < N_ENC)              enc_b[i]                 = f2bf(enc_W[i]);
    else if (i < N_ENC + N_W)   wih_b[i - N_ENC]         = f2bf(W_ih[i - N_ENC]);
    else if (i < N_ENC + 2*N_W) whh_b[i - N_ENC - N_W]   = f2bf(W_hh[i - N_ENC - N_W]);
    else                        fcw_b[i - N_ENC - 2*N_W] = f2bf(fc_W[i - N_ENC - 2*N_W]);
  }
}

// ---------------- stage 1: gi = gather(enc)[x] @ W_ih^T + b_ih  (12800x768) ----------------
__global__ void k_gi(const int* x, const unsigned short* enc_b, const unsigned short* wih_b,
                     const float* b_ih, float* gi) {
  const int lane = threadIdx.x & 31;
  const int wave = threadIdx.x >> 5;
  const int tile = blockIdx.x * 8 + wave;       // 800*48 tiles total
  const int NT = H3 / 16;                        // 48
  const int m0 = (tile / NT) * 16;
  const int n0 = (tile % NT) * 16;
  const int m = m0 + (lane & 15);
  const int token = x[m];
  const __bf16* arow = (const __bf16*)enc_b + (size_t)token * H_DIM;
  const __bf16* B    = (const __bf16*)wih_b;
  f32x8 acc = {};
#pragma unroll
  for (int k0 = 0; k0 < H_DIM; k0 += 32) {
    bf16x16 a = load_a_frag(arow, lane, k0);
    bf16x16 b = load_b_frag(B, H_DIM, n0, lane, k0);
    acc = __builtin_amdgcn_wmma_f32_16x16x32_bf16(false, a, false, b, (short)0, acc, false, false);
  }
  const int n = n0 + (lane & 15);
  const float bias = b_ih[n];
#pragma unroll
  for (int r = 0; r < 8; ++r) {
    const int mr = m0 + r + ((lane >> 4) << 3);
    gi[(size_t)mr * H3 + n] = acc[r] + bias;
  }
}

// ---------------- stage 2: sequential GRU, one persistent workgroup ----------------
__global__ void __launch_bounds__(1024)
k_gru(const float* h0, const unsigned short* whh_b, const float* gi, const float* b_hh,
      float* gh, float* hprev, unsigned short* hprev_b, float* out, float* h_last_out) {
  const int tid  = threadIdx.x;
  const int lane = tid & 31;
  const int wave = tid >> 5;                    // 32 waves
  for (int i = tid; i < U_DIM * H_DIM; i += 1024) {
    const float v = h0[i];
    hprev[i]   = v;
    hprev_b[i] = f2bf(v);
  }
  __threadfence(); __syncthreads();
  const __bf16* Bw = (const __bf16*)whh_b;
  for (int s = 0; s < S_DIM; ++s) {
    const int NT = H3 / 16, MT = U_DIM / 16;    // 48, 8
    for (int tile = wave; tile < MT * NT; tile += 32) {
      const int m0 = (tile / NT) * 16, n0 = (tile % NT) * 16;
      const __bf16* arow = (const __bf16*)hprev_b + (size_t)(m0 + (lane & 15)) * H_DIM;
      f32x8 acc = {};
#pragma unroll
      for (int k0 = 0; k0 < H_DIM; k0 += 32) {
        bf16x16 a = load_a_frag(arow, lane, k0);
        bf16x16 b = load_b_frag(Bw, H_DIM, n0, lane, k0);
        acc = __builtin_amdgcn_wmma_f32_16x16x32_bf16(false, a, false, b, (short)0, acc, false, false);
      }
      const int n = n0 + (lane & 15);
#pragma unroll
      for (int r = 0; r < 8; ++r) {
        const int mr = m0 + r + ((lane >> 4) << 3);
        gh[(size_t)mr * H3 + n] = acc[r];
      }
    }
    __threadfence(); __syncthreads();
    const float* gis = gi + (size_t)s * U_DIM * H3;
    for (int i = tid; i < U_DIM * H_DIM; i += 1024) {
      const int u = i / H_DIM, c = i % H_DIM;
      const float* gu = gis + (size_t)u * H3;
      const float* hu = gh  + (size_t)u * H3;
      const float i_r = gu[c], i_z = gu[c + H_DIM], i_n = gu[c + 2*H_DIM];
      const float h_r = hu[c]           + b_hh[c];
      const float h_z = hu[c +   H_DIM] + b_hh[c +   H_DIM];
      const float h_n = hu[c + 2*H_DIM] + b_hh[c + 2*H_DIM];
      const float r = 1.f / (1.f + __expf(-(i_r + h_r)));
      const float z = 1.f / (1.f + __expf(-(i_z + h_z)));
      const float n = tanhf(i_n + r * h_n);
      const float hp = hprev[i];
      const float hn = (1.f - z) * n + z * hp;
      out[(size_t)s * U_DIM * H_DIM + i] = hn;
      hprev[i]   = hn;
      hprev_b[i] = f2bf(hn);
    }
    __threadfence(); __syncthreads();
  }
  for (int i = tid; i < U_DIM * H_DIM; i += 1024) h_last_out[i] = hprev[i];
}

// ---------------- stage 3: causal spatio-temporal attention -> out_pu (bf16) ----------------
__global__ void k_attn(const float* t, const float* sxy, const float* out,
                       const int* active_user, const float* user_W, unsigned short* opu) {
  const int i = blockIdx.x / U_DIM;
  const int u = blockIdx.x % U_DIM;
  const int h = threadIdx.x;                    // 0..255
  __shared__ float wrow[S_DIM];
  if (h < S_DIM) {
    const int j = h;
    float w = 0.f;
    if (j <= i) {
      const float dt = t[i*U_DIM + u] - t[j*U_DIM + u];
      const float dx = sxy[(i*U_DIM + u)*2 + 0] - sxy[(j*U_DIM + u)*2 + 0];
      const float dy = sxy[(i*U_DIM + u)*2 + 1] - sxy[(j*U_DIM + u)*2 + 1];
      const float ds = sqrtf(dx*dx + dy*dy);
      const float a  = (__cosf(dt * (6.28318530717958647692f / 86400.f)) + 1.f) * 0.5f
                       * __expf(-(dt / 86400.f) * 0.1f);
      const float b  = __expf(-ds * 100.f);
      w = a * b + 1e-10f;
    }
    wrow[j] = w;
  }
  __syncthreads();
  float sumw = 0.f, acc = 0.f;
  for (int j = 0; j <= i; ++j) {
    const float w = wrow[j];
    sumw += w;
    acc  += w * out[((size_t)j * U_DIM + u) * H_DIM + h];
  }
  const size_t base = ((size_t)i * U_DIM + u) * H2;
  opu[base + h]         = f2bf(acc / sumw);
  opu[base + H_DIM + h] = f2bf(user_W[(size_t)active_user[u] * H_DIM + h]);
}

// ---------------- stage 4: final FC (12800x512)@(512x10000) + fc_b ----------------
// Block = 8 waves = 128 rows x 80 cols. B tile (80x32 bf16 = 5KB) is staged in LDS
// by the Tensor Data Mover (double-buffered, overlapped with WMMA compute).
#define FC_BN 80
__global__ void __launch_bounds__(256)
k_fc(const unsigned short* opu, const unsigned short* fcw_b, const float* fc_b, float* y) {
  const int lane = threadIdx.x & 31;
  const int wave = threadIdx.x >> 5;
  const int NB = V_DIM / FC_BN;                  // 125 column-blocks
  const int bm = blockIdx.x / NB;
  const int bn = blockIdx.x % NB;
  const int m0 = bm * 128 + wave * 16;
  const int n_base = bn * FC_BN;

  __shared__ __align__(16) unsigned short Bsh[2][FC_BN * 32];

  const __bf16* arow = (const __bf16*)opu + (size_t)(m0 + (lane & 15)) * H2;
  const __bf16* Bg   = (const __bf16*)fcw_b;

  f32x8 acc[5];
#pragma unroll
  for (int j = 0; j < 5; ++j) acc[j] = {};

  // prologue: TDM-stage k0=0 into buffer 0
  if (wave == 0) {
    tdm_load_2d((unsigned)(unsigned long long)(void*)&Bsh[0][0],
                Bg + (size_t)n_base * H2, H2, V_DIM, 32, FC_BN, H2);
    __builtin_amdgcn_s_wait_tensorcnt(0);
  }
  __syncthreads();

#pragma unroll 1
  for (int kk = 0; kk < H2 / 32; ++kk) {
    const int k0 = kk * 32;
    const int cur = kk & 1;
    if (wave == 0 && kk + 1 < H2 / 32) {        // prefetch next B tile via TDM
      tdm_load_2d((unsigned)(unsigned long long)(void*)&Bsh[cur ^ 1][0],
                  Bg + (size_t)n_base * H2 + (k0 + 32), H2, V_DIM, 32, FC_BN, H2);
    }
    __builtin_prefetch(arow + k0 + 32, 0, 1);   // global_prefetch for next A frag
    bf16x16 a = load_a_frag(arow, lane, k0);
    const int koff = (lane >> 4) << 4;
#pragma unroll
    for (int j = 0; j < 5; ++j) {
      const __bf16* bp = (const __bf16*)&Bsh[cur][(j * 16 + (lane & 15)) * 32 + koff];
      bf16x16 b;
#pragma unroll
      for (int e = 0; e < 16; ++e) b[e] = bp[e];
      acc[j] = __builtin_amdgcn_wmma_f32_16x16x32_bf16(false, a, false, b, (short)0,
                                                       acc[j], false, false);
    }
    if (wave == 0) __builtin_amdgcn_s_wait_tensorcnt(0);
    __syncthreads();
  }

#pragma unroll
  for (int j = 0; j < 5; ++j) {
    const int n = n_base + j * 16 + (lane & 15);
    const float bias = fc_b[n];
#pragma unroll
    for (int r = 0; r < 8; ++r) {
      const int mr = m0 + r + ((lane >> 4) << 3);
      y[(size_t)mr * V_DIM + n] = acc[j][r] + bias;
    }
  }
}

extern "C" void kernel_launch(void* const* d_in, const int* in_sizes, int n_in,
                              void* d_out, int out_size, void* d_ws, size_t ws_size,
                              hipStream_t stream) {
  const int*   x      = (const int*)d_in[0];
  const float* t      = (const float*)d_in[1];
  const float* sxy    = (const float*)d_in[2];
  const float* h0     = (const float*)d_in[5];
  const int*   au     = (const int*)d_in[6];
  const float* enc_W  = (const float*)d_in[7];
  const float* user_W = (const float*)d_in[8];
  const float* W_ih   = (const float*)d_in[9];
  const float* W_hh   = (const float*)d_in[10];
  const float* b_ih   = (const float*)d_in[11];
  const float* b_hh   = (const float*)d_in[12];
  const float* fc_W   = (const float*)d_in[13];
  const float* fc_b   = (const float*)d_in[14];

  float* y      = (float*)d_out;
  float* h_last = y + (size_t)S_DIM * U_DIM * V_DIM;

  char* w = (char*)d_ws;
  unsigned short* enc_b   = (unsigned short*)(w + OFF_ENC);
  unsigned short* wih_b   = (unsigned short*)(w + OFF_WIH);
  unsigned short* whh_b   = (unsigned short*)(w + OFF_WHH);
  unsigned short* fcw_b   = (unsigned short*)(w + OFF_FCW);
  float*          gi      = (float*)(w + OFF_GI);
  float*          gh      = (float*)(w + OFF_GH);
  float*          outbuf  = (float*)(w + OFF_OUT);
  float*          hprev   = (float*)(w + OFF_HPF);
  unsigned short* hprev_b = (unsigned short*)(w + OFF_HPB);
  unsigned short* opu     = (unsigned short*)(w + OFF_OPU);

  k_convert<<<2048, 256, 0, stream>>>(enc_W, W_ih, W_hh, fc_W, enc_b, wih_b, whh_b, fcw_b);
  k_gi<<<(S_DIM * U_DIM / 16) * (H3 / 16) / 8, 256, 0, stream>>>(x, enc_b, wih_b, b_ih, gi);
  k_gru<<<1, 1024, 0, stream>>>(h0, whh_b, gi, b_hh, gh, hprev, hprev_b, outbuf, h_last);
  k_attn<<<S_DIM * U_DIM, 256, 0, stream>>>(t, sxy, outbuf, au, user_W, opu);
  k_fc<<<(S_DIM * U_DIM / 128) * (V_DIM / FC_BN), 256, 0, stream>>>(opu, fcw_b, fc_b, y);
}
